// WindowAttention_30099130810838
// MI455X (gfx1250) — matrix operand order repaired and checked by reading, not compiled, not run on hardware
//
#include <hip/hip_runtime.h>

// Swin window attention, fully fused, one workgroup per window.
// B=2048, N=64 tokens, C=512, 16 heads x 32 dim. bf16 WMMA, f32 accum.
// Weights are pre-converted to bf16 in d_ws by a tiny elementwise kernel.

typedef __attribute__((ext_vector_type(16))) __bf16          v16bf;
typedef __attribute__((ext_vector_type(8)))  float           v8f;
typedef __attribute__((ext_vector_type(16))) unsigned short  v16u;

union V16 { v16u u; v16bf b; };

#define DIMC 512
#define NTOK 64
#define NHC  16
#define NB   2048

// ---- LDS layout (units: ushort/bf16) --------------------------------------
#define RSX   520                 // X / O row stride: 512 + 8 pad
#define XOFF  0                   // X: 64 x 520            (phase 1-2)
#define QOFF  33280               // q: 16 heads x 64 x 40  (phase 2-3)
#define QH    2560
#define KOFF  74240               // k: 16 heads x 64 x 40  (phase 2-3)
#define VOFF  115200              // v^T: 16 heads x 32 x 72(phase 2-3)
#define VH    2304
#define POFF  0                   // P: 16 heads x 64 x 72  (reuses X+Q after barrier)
#define PH    4608
#define PRS   72
#define OOFF  74240               // O: 64 x 520            (reuses K after barriers)
#define SMEM_USHORTS 152064       // 304128 bytes < 320 KB WGP LDS

__device__ __forceinline__ unsigned short f2bf(float f) {
    return __builtin_bit_cast(unsigned short, (__bf16)f);  // native cvt, RNE
}

// A operand (16x32 bf16): lane m = l&15, half h = l>>4;
// elem e -> k = k0+16h+e (contiguous). Two 16B chunks.
__device__ __forceinline__ v16bf load_A(const unsigned short* base, int rowStride,
                                        int m0, int k0, int lane) {
    const int m = m0 + (lane & 15);
    const int h = lane >> 4;
    const unsigned short* p = base + m * rowStride + k0 + 16 * h;
    V16 r;
    #pragma unroll
    for (int e = 0; e < 8; ++e) r.u[e] = p[e];
    #pragma unroll
    for (int e = 0; e < 8; ++e) r.u[8 + e] = p[8 + e];
    return r.b;
}

// B operand (32x16 bf16): lane n = l&15, half h2 = l>>4; elem e -> k = 16*h2+e.
// Pointer = 16 contiguous bf16 for this lane (LDS).
__device__ __forceinline__ v16bf load_B16(const unsigned short* p) {
    V16 r;
    #pragma unroll
    for (int e = 0; e < 16; ++e) r.u[e] = p[e];
    return r.b;
}

// Same but from global bf16 weights: two b128 loads.
__device__ __forceinline__ v16bf load_B_g(const unsigned short* p) {
    union { uint4 q[2]; v16bf b; } r;
    r.q[0] = *(const uint4*)(p);
    r.q[1] = *(const uint4*)(p + 8);
    return r.b;
}

#define WMMA_BF16(A, Bm, C) \
    __builtin_amdgcn_wmma_f32_16x16x32_bf16(false, (A), false, (Bm), (short)0, (C), false, false)

// ---------------------------------------------------------------------------
// Kernel 0: one-time f32 -> bf16 conversion of both weight matrices into ws.
// ---------------------------------------------------------------------------
#define NQ4 ((1536 * 512) / 4)
#define NP4 ((512 * 512) / 4)

__global__ __launch_bounds__(256)
void convert_weights(const float* __restrict__ Wqkv,
                     const float* __restrict__ Wproj,
                     unsigned short* __restrict__ wq,
                     unsigned short* __restrict__ wp) {
    const int idx = blockIdx.x * blockDim.x + threadIdx.x;   // float4 index
    const float4* src;
    unsigned short* dst;
    int i = idx;
    if (idx < NQ4) { src = (const float4*)Wqkv;  dst = wq; }
    else           { src = (const float4*)Wproj; dst = wp; i = idx - NQ4; }
    const float4 f = src[i];
    ushort4 s;
    s.x = f2bf(f.x); s.y = f2bf(f.y); s.z = f2bf(f.z); s.w = f2bf(f.w);
    *(ushort4*)(dst + 4 * i) = s;
}

// ---------------------------------------------------------------------------
// Kernel 1: fused window attention.
// ---------------------------------------------------------------------------
__global__ __launch_bounds__(512, 1)
void swin_window_attn_fused(const float* __restrict__ x,
                            const unsigned short* __restrict__ WqkvBf,
                            const float* __restrict__ bqkv,
                            const unsigned short* __restrict__ WprojBf,
                            const float* __restrict__ bproj,
                            const float* __restrict__ bias_table,
                            float* __restrict__ out) {
    __shared__ __align__(16) unsigned short smem[SMEM_USHORTS];

    const int b    = blockIdx.x;
    const int tid  = threadIdx.x;
    const int w    = tid >> 5;        // wave id, 0..15 (== head in attention)
    const int lane = tid & 31;
    const int n16  = lane & 15;
    const int h2   = lane >> 4;

    // ---------------- Phase 1: X -> LDS as bf16 (b128 stores) ----------------
    const float* xg = x + (size_t)b * (NTOK * DIMC);
    #pragma unroll
    for (int i = 0; i < 8; ++i) {
        const int f8   = tid + i * 512;     // 8-float chunk index, 4096 total
        const int flat = f8 * 8;
        const int row  = flat >> 9;
        const int col  = flat & 511;
        const float4 a = *(const float4*)(xg + flat);
        const float4 c = *(const float4*)(xg + flat + 4);
        union { unsigned short us[8]; uint4 v; } pk;
        pk.us[0] = f2bf(a.x); pk.us[1] = f2bf(a.y);
        pk.us[2] = f2bf(a.z); pk.us[3] = f2bf(a.w);
        pk.us[4] = f2bf(c.x); pk.us[5] = f2bf(c.y);
        pk.us[6] = f2bf(c.z); pk.us[7] = f2bf(c.w);
        *(uint4*)(&smem[row * RSX + col]) = pk.v;
    }
    __syncthreads();

    // ---------------- Phase 2: QKV = X @ Wqkv^T + b ----------------
    // M=64, N=1536, K=512.  16 waves x 6 column tiles each.
    for (int t = 0; t < 6; ++t) {
        const int j0 = (w * 6 + t) * 16;
        const int j  = j0 + n16;                     // output column for this lane
        v8f acc[4];
        v8f z = {};
        #pragma unroll
        for (int mt = 0; mt < 4; ++mt) acc[mt] = z;

        const unsigned short* wrow = WqkvBf + (size_t)j * DIMC + 16 * h2;
        for (int k0 = 0; k0 < DIMC; k0 += 32) {
            __builtin_prefetch(wrow + k0 + 128, 0, 1);
            const v16bf wb = load_B_g(wrow + k0);
            #pragma unroll
            for (int mt = 0; mt < 4; ++mt) {
                const v16bf xa = load_A(smem + XOFF, RSX, mt * 16, k0, lane);
                acc[mt] = WMMA_BF16(xa, wb, acc[mt]);
            }
        }
        const float bq = bqkv[j];
        // tile-uniform routing, forced scalar -> no exec-mask divergence
        const int whichS = __builtin_amdgcn_readfirstlane(j0 >> 9);   // 0=q 1=k 2=v
        const int headS  = __builtin_amdgcn_readfirstlane((j0 >> 5) & 15);
        const int d      = (j0 & 31) + n16;
        if (whichS == 0) {
            const int base = QOFF + headS * QH + d;
            #pragma unroll
            for (int mt = 0; mt < 4; ++mt)
                #pragma unroll
                for (int r = 0; r < 8; ++r) {
                    const int m = mt * 16 + 8 * h2 + r;
                    smem[base + m * 40] = f2bf(acc[mt][r] + bq);
                }
        } else if (whichS == 1) {
            const int base = KOFF + headS * QH + d;
            #pragma unroll
            for (int mt = 0; mt < 4; ++mt)
                #pragma unroll
                for (int r = 0; r < 8; ++r) {
                    const int m = mt * 16 + 8 * h2 + r;
                    smem[base + m * 40] = f2bf(acc[mt][r] + bq);
                }
        } else {
            const int base = VOFF + headS * VH + d * 72;               // v^T
            #pragma unroll
            for (int mt = 0; mt < 4; ++mt)
                #pragma unroll
                for (int r = 0; r < 8; ++r) {
                    const int m = mt * 16 + 8 * h2 + r;
                    smem[base + m] = f2bf(acc[mt][r] + bq);
                }
        }
    }
    __syncthreads();

    // ---------------- Phase 3: attention, one head per wave ----------------
    const int h = __builtin_amdgcn_readfirstlane(w);

    v16bf qa[4];
    #pragma unroll
    for (int mt = 0; mt < 4; ++mt)
        qa[mt] = load_A(smem + QOFF + h * QH, 40, mt * 16, 0, lane);

    v8f s[4][4];
    {
        v8f z = {};
        #pragma unroll
        for (int nt = 0; nt < 4; ++nt) {
            const unsigned short* kp =
                smem + KOFF + h * QH + (nt * 16 + n16) * 40 + 16 * h2;
            const v16bf kb = load_B16(kp);
            #pragma unroll
            for (int mt = 0; mt < 4; ++mt) {
                s[mt][nt] = z;
                s[mt][nt] = WMMA_BF16(qa[mt], kb, s[mt][nt]);
            }
        }
    }

    // scale + relative position bias (Swin rel-pos index computed analytically)
    const float scale = 0.17677669529663687f;   // 32^-0.5
    #pragma unroll
    for (int mt = 0; mt < 4; ++mt) {
        #pragma unroll
        for (int nt = 0; nt < 4; ++nt) {
            const int n  = nt * 16 + n16;
            const int rn = n >> 3, cn = n & 7;
            #pragma unroll
            for (int r = 0; r < 8; ++r) {
                const int m  = mt * 16 + 8 * h2 + r;
                const int rm = m >> 3, cm = m & 7;
                const int idx = (rm - rn + 7) * 15 + (cm - cn + 7);
                s[mt][nt][r] = s[mt][nt][r] * scale + bias_table[idx * NHC + h];
            }
        }
    }

    // row softmax: row m lives in one 16-lane half (C-layout), reduce via shfl_xor
    #pragma unroll
    for (int mt = 0; mt < 4; ++mt) {
        #pragma unroll
        for (int r = 0; r < 8; ++r) {
            float mx = -3.0e38f;
            #pragma unroll
            for (int nt = 0; nt < 4; ++nt) mx = fmaxf(mx, s[mt][nt][r]);
            #pragma unroll
            for (int msk = 1; msk < 16; msk <<= 1)
                mx = fmaxf(mx, __shfl_xor(mx, msk, 32));
            float sum = 0.f;
            #pragma unroll
            for (int nt = 0; nt < 4; ++nt) {
                const float p = __expf(s[mt][nt][r] - mx);
                s[mt][nt][r] = p;
                sum += p;
            }
            #pragma unroll
            for (int msk = 1; msk < 16; msk <<= 1)
                sum += __shfl_xor(sum, msk, 32);
            const float rinv = 1.0f / sum;
            #pragma unroll
            for (int nt = 0; nt < 4; ++nt) s[mt][nt][r] *= rinv;
        }
    }
    __syncthreads();   // X and Q regions now dead -> safe to overwrite with P

    #pragma unroll
    for (int mt = 0; mt < 4; ++mt)
        #pragma unroll
        for (int nt = 0; nt < 4; ++nt)
            #pragma unroll
            for (int r = 0; r < 8; ++r) {
                const int m = mt * 16 + 8 * h2 + r;
                const int n = nt * 16 + n16;
                smem[POFF + h * PH + m * PRS + n] = f2bf(s[mt][nt][r]);
            }
    __syncthreads();

    // P @ V : M=64, N=32, K=64
    v8f o[4][2];
    {
        v8f z = {};
        #pragma unroll
        for (int mt = 0; mt < 4; ++mt)
            #pragma unroll
            for (int dt = 0; dt < 2; ++dt) o[mt][dt] = z;
    }
    #pragma unroll
    for (int kt = 0; kt < 2; ++kt) {
        v16bf vb[2];
        #pragma unroll
        for (int dt = 0; dt < 2; ++dt)
            vb[dt] = load_B16(smem + VOFF + h * VH + (dt * 16 + n16) * 72 +
                              kt * 32 + 16 * h2);
        #pragma unroll
        for (int mt = 0; mt < 4; ++mt) {
            const v16bf pa = load_A(smem + POFF + h * PH, PRS, mt * 16, kt * 32, lane);
            #pragma unroll
            for (int dt = 0; dt < 2; ++dt)
                o[mt][dt] = WMMA_BF16(pa, vb[dt], o[mt][dt]);
        }
    }
    // store O (K region is dead after the softmax barrier)
    #pragma unroll
    for (int mt = 0; mt < 4; ++mt)
        #pragma unroll
        for (int dt = 0; dt < 2; ++dt)
            #pragma unroll
            for (int r = 0; r < 8; ++r) {
                const int m   = mt * 16 + 8 * h2 + r;
                const int col = h * 32 + dt * 16 + n16;
                smem[OOFF + m * RSX + col] = f2bf(o[mt][dt][r]);
            }
    __syncthreads();

    // ---------------- Phase 4: out = O @ Wproj^T + b ----------------
    float* og = out + (size_t)b * (NTOK * DIMC);
    for (int t = 0; t < 2; ++t) {
        const int j0 = (w * 2 + t) * 16;
        const int jj = j0 + n16;
        v8f acc[4];
        v8f z = {};
        #pragma unroll
        for (int mt = 0; mt < 4; ++mt) acc[mt] = z;

        const unsigned short* wrow = WprojBf + (size_t)jj * DIMC + 16 * h2;
        for (int k0 = 0; k0 < DIMC; k0 += 32) {
            __builtin_prefetch(wrow + k0 + 128, 0, 1);
            const v16bf wb = load_B_g(wrow + k0);
            #pragma unroll
            for (int mt = 0; mt < 4; ++mt) {
                const v16bf oa = load_A(smem + OOFF, RSX, mt * 16, k0, lane);
                acc[mt] = WMMA_BF16(oa, wb, acc[mt]);
            }
        }
        const float bp = bproj[jj];
        #pragma unroll
        for (int mt = 0; mt < 4; ++mt) {
            #pragma unroll
            for (int r = 0; r < 8; ++r) {
                const int m = mt * 16 + 8 * h2 + r;
                og[(size_t)m * DIMC + jj] = acc[mt][r] + bp;
            }
        }
    }
}

extern "C" void kernel_launch(void* const* d_in, const int* in_sizes, int n_in,
                              void* d_out, int out_size, void* d_ws, size_t ws_size,
                              hipStream_t stream) {
    (void)in_sizes; (void)n_in; (void)out_size; (void)ws_size;
    const float* x     = (const float*)d_in[0];
    const float* Wqkv  = (const float*)d_in[1];
    const float* bqkv  = (const float*)d_in[2];
    const float* Wproj = (const float*)d_in[3];
    const float* bproj = (const float*)d_in[4];
    const float* bt    = (const float*)d_in[5];

    unsigned short* wqBf = (unsigned short*)d_ws;                    // 1536*512 bf16
    unsigned short* wpBf = wqBf + (size_t)1536 * 512;                //  512*512 bf16

    convert_weights<<<dim3((NQ4 + NP4) / 256), dim3(256), 0, stream>>>(
        Wqkv, Wproj, wqBf, wpBf);
    swin_window_attn_fused<<<dim3(NB), dim3(512), 0, stream>>>(
        x, wqBf, bqkv, wpBf, bproj, bt, (float*)d_out);
}
